// SparseMOE_66606352826417
// MI455X (gfx1250) — matrix operand order repaired and checked by reading, not compiled, not run on hardware
//
#include <hip/hip_runtime.h>
#include <hip/hip_bf16.h>
#include <math.h>

// ---------------------------------------------------------------------------
// MoE forward for MI455X (gfx1250, wave32, WMMA).
// T=8192 tokens, D=1024, H=4096, E=8, top-2. bf16 WMMA w/ fp32 accum.
// ---------------------------------------------------------------------------

#define T_TOK 8192
#define DIM   1024
#define HID   4096
#define NE    8
#define NSLOT (2 * T_TOK)   // total selected rows = exactly 2T

typedef __attribute__((ext_vector_type(16))) __bf16        v16bf;
typedef __attribute__((ext_vector_type(8)))  float         v8f;
typedef __attribute__((ext_vector_type(4)))  unsigned int  u32x4;
typedef __attribute__((ext_vector_type(4)))  float         f32x4;

union Frag { v16bf v; u32x4 u[2]; };

static __device__ __forceinline__ unsigned int f2bf(float f) {
  unsigned int u = __float_as_uint(f);
  return (u + 0x7FFFu + ((u >> 16) & 1u)) >> 16;   // RNE
}

static __device__ __forceinline__ float gelu_f(float v) {
  return 0.5f * v * (1.0f + erff(v * 0.70710678118654752440f));
}

// ---------------------------------------------------------------------------
// ws counter block layout (ints): [0..7]=counts, [8..15]=cursor, [16..23]=offsets
// ---------------------------------------------------------------------------

__global__ void moe_zero(int* ctrs) {
  if (threadIdx.x < NE) ctrs[threadIdx.x] = 0;
}

__global__ void moe_scan(int* ctrs) {
  if (threadIdx.x == 0) {
    int s = 0;
    for (int e = 0; e < NE; ++e) {
      ctrs[16 + e] = s;   // offsets
      ctrs[8 + e]  = s;   // cursor
      s += ctrs[e];
    }
  }
}

// x (fp32) -> xb (bf16), 8 elems / thread
__global__ __launch_bounds__(256) void moe_cvt_x(const float* __restrict__ x,
                                                 unsigned short* __restrict__ xb) {
  size_t i = (size_t)(blockIdx.x * 256 + threadIdx.x) * 8;
  f32x4 f0 = *(const f32x4*)(x + i);
  f32x4 f1 = *(const f32x4*)(x + i + 4);
  u32x4 o;
  o[0] = f2bf(f0[0]) | (f2bf(f0[1]) << 16);
  o[1] = f2bf(f0[2]) | (f2bf(f0[3]) << 16);
  o[2] = f2bf(f1[0]) | (f2bf(f1[1]) << 16);
  o[3] = f2bf(f1[2]) | (f2bf(f1[3]) << 16);
  *(u32x4*)(xb + i) = o;
}

// w [E][R][C] fp32  ->  wt [E][C][R] bf16   (32x32 LDS tiles, coalesced both ways)
__global__ __launch_bounds__(256) void moe_transpose(const float* __restrict__ w,
                                                     unsigned short* __restrict__ wt,
                                                     int R, int C) {
  __shared__ float tile[32][33];
  const int e  = blockIdx.z;
  const int r0 = blockIdx.y * 32;
  const int c0 = blockIdx.x * 32;
  const int tx = threadIdx.x & 31;
  const int ty = threadIdx.x >> 5;             // 0..7
  const float* src = w + ((size_t)e * R) * C;
#pragma unroll
  for (int j = 0; j < 4; ++j)
    tile[ty + 8 * j][tx] = src[(size_t)(r0 + ty + 8 * j) * C + (c0 + tx)];
  __syncthreads();
  unsigned short* dst = wt + ((size_t)e * C) * R;
#pragma unroll
  for (int j = 0; j < 4; ++j)
    dst[(size_t)(c0 + ty + 8 * j) * R + (r0 + tx)] =
        (unsigned short)f2bf(tile[tx][ty + 8 * j]);
}

// Router: one wave per token. logits fp32 -> d_out tail; top-2 renorm weights.
__global__ __launch_bounds__(256) void moe_router(const float* __restrict__ x,
                                                  const float* __restrict__ gw,
                                                  const float* __restrict__ gb,
                                                  float* __restrict__ logitsOut,
                                                  int* __restrict__ ctrs,
                                                  int* __restrict__ tokE,
                                                  float* __restrict__ tokW) {
  const int lane = threadIdx.x & 31;
  const int t = blockIdx.x * 8 + (threadIdx.x >> 5);
  float acc[NE];
#pragma unroll
  for (int e = 0; e < NE; ++e) acc[e] = 0.0f;
  const float* xr = x + (size_t)t * DIM;
  for (int i = 0; i < DIM / 32; ++i) {
    const int d = i * 32 + lane;
    const float xv = xr[d];
    const f32x4* g = (const f32x4*)(gw + (size_t)d * NE);
    f32x4 g0 = g[0], g1 = g[1];
    acc[0] += xv * g0[0]; acc[1] += xv * g0[1];
    acc[2] += xv * g0[2]; acc[3] += xv * g0[3];
    acc[4] += xv * g1[0]; acc[5] += xv * g1[1];
    acc[6] += xv * g1[2]; acc[7] += xv * g1[3];
  }
#pragma unroll
  for (int e = 0; e < NE; ++e)
#pragma unroll
    for (int s = 16; s > 0; s >>= 1) acc[e] += __shfl_xor(acc[e], s, 32);

  if (lane == 0) {
    float l[NE];
#pragma unroll
    for (int e = 0; e < NE; ++e) {
      l[e] = acc[e] + gb[e];
      logitsOut[(size_t)t * NE + e] = l[e];
    }
    int e0 = 0;
#pragma unroll
    for (int e = 1; e < NE; ++e) if (l[e] > l[e0]) e0 = e;
    int e1 = (e0 == 0) ? 1 : 0;
#pragma unroll
    for (int e = 0; e < NE; ++e) if (e != e0 && l[e] > l[e1]) e1 = e;
    // renormalized top-2 softmax weights (math-identical to softmax->topk->renorm)
    const float q  = expf(l[e1] - l[e0]);
    const float w0 = 1.0f / (1.0f + q);
    const float w1 = q * w0;
    tokE[2 * t] = e0;  tokW[2 * t] = w0;
    tokE[2 * t + 1] = e1;  tokW[2 * t + 1] = w1;
    atomicAdd(&ctrs[e0], 1);
    atomicAdd(&ctrs[e1], 1);
  }
}

// Build compact per-expert row segments.
__global__ __launch_bounds__(256) void moe_build(const int* __restrict__ tokE,
                                                 const float* __restrict__ tokW,
                                                 int* __restrict__ ctrs,
                                                 int* __restrict__ rowTok,
                                                 float* __restrict__ rowW,
                                                 int* __restrict__ tok2row) {
  const int t = blockIdx.x * 256 + threadIdx.x;
#pragma unroll
  for (int k = 0; k < 2; ++k) {
    const int e = tokE[2 * t + k];
    const int pos = atomicAdd(&ctrs[8 + e], 1);
    rowTok[pos] = t;
    rowW[pos] = tokW[2 * t + k];
    tok2row[2 * t + k] = pos;
  }
}

// ---------------------------------------------------------------------------
// Stage-1: mid[row, H] = gelu( xb[tok(row)] @ w1[e] + b1[e] ),  bf16 output.
// WG tile 128x128, 8 waves (4Mx2N), wave tile 32x64 = 2x4 wmma frags, K-step 32.
// ---------------------------------------------------------------------------
__global__ __launch_bounds__(256) void moe_stage1(const unsigned short* __restrict__ xb,
                                                  const unsigned short* __restrict__ w1t,
                                                  const float* __restrict__ b1,
                                                  const int* __restrict__ ctrs,
                                                  const int* __restrict__ rowTok,
                                                  unsigned short* __restrict__ mid) {
  const int e = blockIdx.z;
  const int cnt = ctrs[e];
  const int mtile = blockIdx.y;
  if (mtile * 128 >= cnt) return;
  const int off = ctrs[16 + e];
  const int ntile = blockIdx.x;

  __shared__ unsigned short As[2][128][32];
  __shared__ unsigned short Bs[2][128][32];

  const int tid = threadIdx.x;
  const int lane = tid & 31;
  const int wm = (tid >> 5) & 3;
  const int wn = tid >> 7;

  const int srow  = tid >> 1;
  const int shalf = (tid & 1) * 16;

  const int rIdx = mtile * 128 + srow;
  const int tok  = (rIdx < cnt) ? rowTok[off + rIdx] : 0;
  const unsigned short* aSrc = xb + (size_t)tok * DIM + shalf;
  const unsigned short* bSrc =
      w1t + ((size_t)e * HID + (size_t)ntile * 128 + srow) * DIM + shalf;

  v8f acc[2][4];
  const v8f vzero = {};
#pragma unroll
  for (int i = 0; i < 2; ++i)
#pragma unroll
    for (int j = 0; j < 4; ++j) acc[i][j] = vzero;

  u32x4 ar0, ar1, br0, br1;
  ar0 = *(const u32x4*)(aSrc);     ar1 = *(const u32x4*)(aSrc + 8);
  br0 = *(const u32x4*)(bSrc);     br1 = *(const u32x4*)(bSrc + 8);
  *(u32x4*)&As[0][srow][shalf] = ar0;  *(u32x4*)&As[0][srow][shalf + 8] = ar1;
  *(u32x4*)&Bs[0][srow][shalf] = br0;  *(u32x4*)&Bs[0][srow][shalf + 8] = br1;
  __syncthreads();

  const int laneLo = lane & 15;
  const int laneHi = lane >> 4;
  const int KT = DIM / 32;
  int buf = 0;
  for (int kt = 0; kt < KT; ++kt) {
    if (kt + 1 < KT) {
      const size_t kg = (size_t)(kt + 1) * 32;
      ar0 = *(const u32x4*)(aSrc + kg);  ar1 = *(const u32x4*)(aSrc + kg + 8);
      br0 = *(const u32x4*)(bSrc + kg);  br1 = *(const u32x4*)(bSrc + kg + 8);
    }
    Frag a[2], b[4];
#pragma unroll
    for (int fm = 0; fm < 2; ++fm) {
      const int m = wm * 32 + fm * 16 + laneLo;
      const int ko = laneHi * 8;
      a[fm].u[0] = *(const u32x4*)&As[buf][m][ko];
      a[fm].u[1] = *(const u32x4*)&As[buf][m][16 + ko];
    }
#pragma unroll
    for (int fn = 0; fn < 4; ++fn) {
      const int n = wn * 64 + fn * 16 + laneLo;
      const int kb = laneHi * 8;
      b[fn].u[0] = *(const u32x4*)&Bs[buf][n][kb];
      b[fn].u[1] = *(const u32x4*)&Bs[buf][n][16 + kb];
    }
#pragma unroll
    for (int fm = 0; fm < 2; ++fm)
#pragma unroll
      for (int fn = 0; fn < 4; ++fn)
        acc[fm][fn] = __builtin_amdgcn_wmma_f32_16x16x32_bf16(
            false, a[fm].v, false, b[fn].v, (short)0, acc[fm][fn], false, false);

    if (kt + 1 < KT) {
      const int nb = buf ^ 1;
      *(u32x4*)&As[nb][srow][shalf] = ar0;  *(u32x4*)&As[nb][srow][shalf + 8] = ar1;
      *(u32x4*)&Bs[nb][srow][shalf] = br0;  *(u32x4*)&Bs[nb][srow][shalf + 8] = br1;
      __syncthreads();
      buf = nb;
    }
  }

  // epilogue: + b1, gelu, -> bf16 mid
#pragma unroll
  for (int fn = 0; fn < 4; ++fn) {
    const int n = ntile * 128 + wn * 64 + fn * 16 + laneLo;
    const float bias = b1[(size_t)e * HID + n];
#pragma unroll
    for (int fm = 0; fm < 2; ++fm) {
#pragma unroll
      for (int r = 0; r < 8; ++r) {
        const int mIdx = mtile * 128 + wm * 32 + fm * 16 + r + 8 * laneHi;
        if (mIdx < cnt) {
          const float v = gelu_f(acc[fm][fn][r] + bias);
          mid[(size_t)(off + mIdx) * HID + n] = (unsigned short)f2bf(v);
        }
      }
    }
  }
}

// ---------------------------------------------------------------------------
// Stage-2: contrib[row, D] = rowW[row] * ( mid[row] @ w2[e] + b2[e] ), fp32.
// ---------------------------------------------------------------------------
__global__ __launch_bounds__(256) void moe_stage2(const unsigned short* __restrict__ mid,
                                                  const unsigned short* __restrict__ w2t,
                                                  const float* __restrict__ b2,
                                                  const int* __restrict__ ctrs,
                                                  const float* __restrict__ rowW,
                                                  float* __restrict__ contrib) {
  const int e = blockIdx.z;
  const int cnt = ctrs[e];
  const int mtile = blockIdx.y;
  if (mtile * 128 >= cnt) return;
  const int off = ctrs[16 + e];
  const int ntile = blockIdx.x;

  __shared__ unsigned short As[2][128][32];
  __shared__ unsigned short Bs[2][128][32];

  const int tid = threadIdx.x;
  const int lane = tid & 31;
  const int wm = (tid >> 5) & 3;
  const int wn = tid >> 7;

  const int srow  = tid >> 1;
  const int shalf = (tid & 1) * 16;

  int slot = off + mtile * 128 + srow;
  if (slot > NSLOT - 1) slot = NSLOT - 1;           // clamp padded rows
  const unsigned short* aSrc = mid + (size_t)slot * HID + shalf;
  const unsigned short* bSrc =
      w2t + ((size_t)e * DIM + (size_t)ntile * 128 + srow) * HID + shalf;

  v8f acc[2][4];
  const v8f vzero = {};
#pragma unroll
  for (int i = 0; i < 2; ++i)
#pragma unroll
    for (int j = 0; j < 4; ++j) acc[i][j] = vzero;

  u32x4 ar0, ar1, br0, br1;
  ar0 = *(const u32x4*)(aSrc);     ar1 = *(const u32x4*)(aSrc + 8);
  br0 = *(const u32x4*)(bSrc);     br1 = *(const u32x4*)(bSrc + 8);
  *(u32x4*)&As[0][srow][shalf] = ar0;  *(u32x4*)&As[0][srow][shalf + 8] = ar1;
  *(u32x4*)&Bs[0][srow][shalf] = br0;  *(u32x4*)&Bs[0][srow][shalf + 8] = br1;
  __syncthreads();

  const int laneLo = lane & 15;
  const int laneHi = lane >> 4;
  const int KT = HID / 32;
  int buf = 0;
  for (int kt = 0; kt < KT; ++kt) {
    if (kt + 1 < KT) {
      const size_t kg = (size_t)(kt + 1) * 32;
      ar0 = *(const u32x4*)(aSrc + kg);  ar1 = *(const u32x4*)(aSrc + kg + 8);
      br0 = *(const u32x4*)(bSrc + kg);  br1 = *(const u32x4*)(bSrc + kg + 8);
    }
    Frag a[2], b[4];
#pragma unroll
    for (int fm = 0; fm < 2; ++fm) {
      const int m = wm * 32 + fm * 16 + laneLo;
      const int ko = laneHi * 8;
      a[fm].u[0] = *(const u32x4*)&As[buf][m][ko];
      a[fm].u[1] = *(const u32x4*)&As[buf][m][16 + ko];
    }
#pragma unroll
    for (int fn = 0; fn < 4; ++fn) {
      const int n = wn * 64 + fn * 16 + laneLo;
      const int kb = laneHi * 8;
      b[fn].u[0] = *(const u32x4*)&Bs[buf][n][kb];
      b[fn].u[1] = *(const u32x4*)&Bs[buf][n][16 + kb];
    }
#pragma unroll
    for (int fm = 0; fm < 2; ++fm)
#pragma unroll
      for (int fn = 0; fn < 4; ++fn)
        acc[fm][fn] = __builtin_amdgcn_wmma_f32_16x16x32_bf16(
            false, a[fm].v, false, b[fn].v, (short)0, acc[fm][fn], false, false);

    if (kt + 1 < KT) {
      const int nb = buf ^ 1;
      *(u32x4*)&As[nb][srow][shalf] = ar0;  *(u32x4*)&As[nb][srow][shalf + 8] = ar1;
      *(u32x4*)&Bs[nb][srow][shalf] = br0;  *(u32x4*)&Bs[nb][srow][shalf + 8] = br1;
      __syncthreads();
      buf = nb;
    }
  }

  // epilogue: + b2, * router weight, fp32 contrib
#pragma unroll
  for (int fn = 0; fn < 4; ++fn) {
    const int n = ntile * 128 + wn * 64 + fn * 16 + laneLo;
    const float bias = b2[(size_t)e * DIM + n];
#pragma unroll
    for (int fm = 0; fm < 2; ++fm) {
#pragma unroll
      for (int r = 0; r < 8; ++r) {
        const int mIdx = mtile * 128 + wm * 32 + fm * 16 + r + 8 * laneHi;
        if (mIdx < cnt) {
          const float w = rowW[off + mIdx];
          contrib[(size_t)(off + mIdx) * DIM + n] = (acc[fm][fn][r] + bias) * w;
        }
      }
    }
  }
}

// out[t] = contrib[row0(t)] + contrib[row1(t)]   (no fp atomics -> deterministic)
__global__ __launch_bounds__(256) void moe_combine(const float* __restrict__ contrib,
                                                   const int* __restrict__ tok2row,
                                                   float* __restrict__ out) {
  const int i = blockIdx.x * 256 + threadIdx.x;   // T * (D/4) threads
  const int t = i >> 8;                           // D/4 = 256
  const int c = i & 255;
  const int r0 = tok2row[2 * t];
  const int r1 = tok2row[2 * t + 1];
  const f32x4* c4 = (const f32x4*)contrib;
  f32x4 a = c4[(size_t)r0 * 256 + c];
  f32x4 b = c4[(size_t)r1 * 256 + c];
  ((f32x4*)out)[(size_t)t * 256 + c] = a + b;
}

// ---------------------------------------------------------------------------

extern "C" void kernel_launch(void* const* d_in, const int* in_sizes, int n_in,
                              void* d_out, int out_size, void* d_ws, size_t ws_size,
                              hipStream_t stream) {
  const float* x      = (const float*)d_in[0];
  const float* gate_w = (const float*)d_in[1];
  const float* gate_b = (const float*)d_in[2];
  const float* w1     = (const float*)d_in[3];
  const float* b1     = (const float*)d_in[4];
  const float* w2     = (const float*)d_in[5];
  const float* b2     = (const float*)d_in[6];

  float* out       = (float*)d_out;
  float* logitsOut = out + (size_t)T_TOK * DIM;

  // workspace carve-up (256B aligned)
  char* p = (char*)d_ws;
  auto carve = [&](size_t bytes) -> char* {
    char* r = p;
    p += (bytes + 255) & ~(size_t)255;
    return r;
  };
  int*            ctrs    = (int*)carve(24 * sizeof(int));
  int*            tokE    = (int*)carve((size_t)NSLOT * sizeof(int));
  float*          tokW    = (float*)carve((size_t)NSLOT * sizeof(float));
  int*            rowTok  = (int*)carve((size_t)NSLOT * sizeof(int));
  float*          rowW    = (float*)carve((size_t)NSLOT * sizeof(float));
  int*            tok2row = (int*)carve((size_t)NSLOT * sizeof(int));
  unsigned short* xb      = (unsigned short*)carve((size_t)T_TOK * DIM * 2);
  unsigned short* w1t     = (unsigned short*)carve((size_t)NE * DIM * HID * 2);
  unsigned short* w2t     = (unsigned short*)carve((size_t)NE * HID * DIM * 2);
  unsigned short* mid     = (unsigned short*)carve((size_t)NSLOT * HID * 2);
  float*          contrib = (float*)carve((size_t)NSLOT * DIM * sizeof(float));

  moe_zero<<<1, 32, 0, stream>>>(ctrs);
  moe_cvt_x<<<(T_TOK * DIM / 8) / 256, 256, 0, stream>>>(x, xb);
  moe_transpose<<<dim3(HID / 32, DIM / 32, NE), 256, 0, stream>>>(w1, w1t, DIM, HID);
  moe_transpose<<<dim3(DIM / 32, HID / 32, NE), 256, 0, stream>>>(w2, w2t, HID, DIM);
  moe_router<<<T_TOK / 8, 256, 0, stream>>>(x, gate_w, gate_b, logitsOut, ctrs, tokE, tokW);
  moe_scan<<<1, 32, 0, stream>>>(ctrs);
  moe_build<<<T_TOK / 256, 256, 0, stream>>>(tokE, tokW, ctrs, rowTok, rowW, tok2row);
  moe_stage1<<<dim3(HID / 128, T_TOK / 128, NE), 256, 0, stream>>>(xb, w1t, b1, ctrs, rowTok, mid);
  moe_stage2<<<dim3(DIM / 128, T_TOK / 128, NE), 256, 0, stream>>>(mid, w2t, b2, ctrs, rowW, contrib);
  moe_combine<<<(T_TOK * (DIM / 4)) / 256, 256, 0, stream>>>(contrib, tok2row, out);
}
